// TransformerBlock_16423954940132
// MI455X (gfx1250) — compile-verified, matrix operand order checked by reading, not run on hardware
//
#include <hip/hip_runtime.h>
#include <hip/hip_bf16.h>
#include <math.h>

// Problem constants (from reference)
static constexpr int kB   = 4;
static constexpr int kN   = 1024;
static constexpr int kD   = 1024;
static constexpr int kH   = 16;
static constexpr int kE   = 12;
static constexpr int kHD  = 64;
static constexpr int kFF  = 4096;   // FF * D
static constexpr int kT   = kB * kN;

typedef __bf16 bf16;
typedef __attribute__((ext_vector_type(16))) __bf16 v16bf;
typedef __attribute__((ext_vector_type(8)))  __bf16 v8bf;
typedef __attribute__((ext_vector_type(4)))  __bf16 v4bf;
typedef __attribute__((ext_vector_type(8)))  float  v8f;
typedef __attribute__((ext_vector_type(4)))  float  v4f;

__device__ inline v8f zero8() {
  v8f z;
#pragma unroll
  for (int i = 0; i < 8; ++i) z[i] = 0.0f;
  return z;
}

// Load a 16x32 bf16 A/B fragment for v_wmma_f32_16x16x32_bf16.
// Per ISA 7.12.2: per lane, elements 0..7 = K kl..kl+7, elements 8..15 = K 16+kl..16+kl+7,
// both contiguous 16-byte chunks (kl = 8*(lane>=16)).  p must already include row*stride + kl.
__device__ inline v16bf load_frag_bf16(const bf16* p) {
  v8bf lo = *(const v8bf*)p;
  v8bf hh = *(const v8bf*)(p + 16);
  v16bf a;
#pragma unroll
  for (int j = 0; j < 8; ++j) { a[j] = lo[j]; a[8 + j] = hh[j]; }
  return a;
}

// fp32 -> bf16 fragment conversion (fallback path only).
__device__ inline v16bf cvt_frag_f32(const float* p) {
  v4f w0 = *(const v4f*)p;
  v4f w1 = *(const v4f*)(p + 4);
  v4f w2 = *(const v4f*)(p + 16);
  v4f w3 = *(const v4f*)(p + 20);
  v16bf b;
#pragma unroll
  for (int j = 0; j < 4; ++j) {
    b[j]      = (bf16)w0[j];
    b[4 + j]  = (bf16)w1[j];
    b[8 + j]  = (bf16)w2[j];
    b[12 + j] = (bf16)w3[j];
  }
  return b;
}

// Async 16B global -> LDS copy (CDNA5, ASYNCcnt-tracked).
__device__ inline void async_copy_b128(unsigned lds_off, const bf16* g) {
  unsigned long long ga = (unsigned long long)(size_t)g;
  asm volatile("global_load_async_to_lds_b128 %0, %1, off"
               :: "v"(lds_off), "v"(ga) : "memory");
}
__device__ inline void wait_async0() {
  asm volatile("s_wait_asynccnt 0" ::: "memory");
}

// ---------------- fp32 -> bf16 bulk weight conversion ----------------
__global__ void cvt_f32_bf16_kernel(const float* __restrict__ in, bf16* __restrict__ out, int n4) {
  int i = blockIdx.x * 256 + threadIdx.x;
  if (i < n4) {
    v4f v = *(const v4f*)(in + (size_t)i * 4);
    v4bf o;
#pragma unroll
    for (int j = 0; j < 4; ++j) o[j] = (bf16)v[j];
    *(v4bf*)(out + (size_t)i * 4) = o;
  }
}

// ---------------- LayerNorm (fp32 in, bf16 out) ----------------
__global__ void ln_bf16_kernel(const float* __restrict__ x, const float* __restrict__ g,
                               const float* __restrict__ be, bf16* __restrict__ out) {
  __shared__ float s1[256], s2[256];
  const int row = blockIdx.x, t = threadIdx.x;
  const float* xr = x + (size_t)row * kD;
  float a = 0.f, b = 0.f;
#pragma unroll
  for (int i = 0; i < kD / 256; ++i) { float v = xr[t + i * 256]; a += v; b += v * v; }
  s1[t] = a; s2[t] = b;
  __syncthreads();
  for (int off = 128; off > 0; off >>= 1) {
    if (t < off) { s1[t] += s1[t + off]; s2[t] += s2[t + off]; }
    __syncthreads();
  }
  const float mean = s1[0] / kD;
  const float var  = s2[0] / kD - mean * mean;
  const float rstd = rsqrtf(var + 1e-5f);
  bf16* orow = out + (size_t)row * kD;
#pragma unroll
  for (int i = 0; i < kD / 256; ++i) {
    int c = t + i * 256;
    orow[c] = (bf16)((xr[c] - mean) * rstd * g[c] + be[c]);
  }
}

// ---------------- Router: logits, softmax, top-2, combine weights + lb stats ----------------
__global__ void router_kernel(const bf16* __restrict__ xn, const float* __restrict__ Wr,
                              float* __restrict__ combine, float* __restrict__ counts,
                              float* __restrict__ psum) {
  __shared__ float part[kE][128];
  __shared__ float logit_s[kE];
  const int row = blockIdx.x, t = threadIdx.x;
  const bf16* xr = xn + (size_t)row * kD;
  float acc[kE];
#pragma unroll
  for (int e = 0; e < kE; ++e) acc[e] = 0.f;
  for (int d = t; d < kD; d += 128) {
    float xv = (float)xr[d];
#pragma unroll
    for (int e = 0; e < kE; ++e) acc[e] += xv * Wr[(size_t)e * kD + d];
  }
#pragma unroll
  for (int e = 0; e < kE; ++e) part[e][t] = acc[e];
  __syncthreads();
  if (t < kE) {
    float s = 0.f;
    for (int i = 0; i < 128; ++i) s += part[t][i];
    logit_s[t] = s;
  }
  __syncthreads();
  if (t == 0) {
    float mx = -1e30f;
#pragma unroll
    for (int e = 0; e < kE; ++e) mx = fmaxf(mx, logit_s[e]);
    float p[kE]; float sum = 0.f;
#pragma unroll
    for (int e = 0; e < kE; ++e) { p[e] = __expf(logit_s[e] - mx); sum += p[e]; }
#pragma unroll
    for (int e = 0; e < kE; ++e) p[e] /= sum;
    int i0 = 0;
#pragma unroll
    for (int e = 1; e < kE; ++e) if (p[e] > p[i0]) i0 = e;
    int i1 = (i0 == 0) ? 1 : 0;
#pragma unroll
    for (int e = 0; e < kE; ++e) if (e != i0 && p[e] > p[i1]) i1 = e;
    float inv = 1.f / (p[i0] + p[i1] + 1e-6f);
    float* crow = combine + (size_t)row * kE;
#pragma unroll
    for (int e = 0; e < kE; ++e) crow[e] = 0.f;
    crow[i0] = p[i0] * inv;
    crow[i1] = p[i1] * inv;
    atomicAdd(&counts[i0], 1.f);
    atomicAdd(&counts[i1], 1.f);
#pragma unroll
    for (int e = 0; e < kE; ++e) atomicAdd(&psum[e], p[e]);
  }
}

// ---------------- WMMA GEMM, bf16 weights, async-LDS staged, double buffered ----------------
// Block: 256 threads = 8 waves; wave grid 4(M) x 2(N); wave tile 16x64 -> block tile 64x128.
// Per k-step the 8 KB W tile [128 cols x 32 k] is staged in LDS by 2x16B async copies per
// thread; the 4 M-waves then share it (4x less global weight traffic).
template <bool MOE, bool BIAS, bool GELU, bool RES, bool OUT_BF16>
__global__ void __launch_bounds__(256)
wmma_gemm_bf16w_kernel(const bf16* __restrict__ X, const bf16* __restrict__ Wb,
                       const float* __restrict__ combine, const float* __restrict__ bias,
                       const float* __restrict__ resid, float* __restrict__ Yf,
                       bf16* __restrict__ Ybf, int Din, int Dout, int nExp) {
  __shared__ __align__(16) bf16 wtile[2][128 * 32];
  const int tid  = threadIdx.x;
  const int lane = tid & 31;
  const int wid  = tid >> 5;
  const int rowBase = blockIdx.x * 64 + (wid & 3) * 16;
  const int col0    = blockIdx.y * 128;
  const int colSub  = (wid >> 2) * 64;       // this wave's 64-col slice within the tile
  const int r  = lane & 15;
  const int kl = (lane >> 4) << 3;
  const int hi = kl;

  const int ksteps = Din / 32;
  const int NE     = MOE ? nExp : 1;
  const int ntiles = NE * ksteps;

  auto issue_copy = [&](int tix, int buf) {
    const int e  = MOE ? (tix / ksteps) : 0;
    const int k0 = (tix - e * ksteps) * 32;
    const bf16* Wbase = Wb + (size_t)e * (size_t)Dout * (size_t)Din + k0;
#pragma unroll
    for (int c = 0; c < 2; ++c) {
      const int ch = tid + c * 256;           // 512 chunks of 16B = 8KB tile
      const int wrow = ch >> 2, seg = ch & 3; // row = output col in tile, seg = 8-elem group
      const bf16* g = Wbase + (size_t)(col0 + wrow) * Din + seg * 8;
      unsigned lds_off = (unsigned)(size_t)(&wtile[buf][ch * 8]);
      async_copy_b128(lds_off, g);
    }
  };

  v8f out[4];
#pragma unroll
  for (int s = 0; s < 4; ++s) out[s] = zero8();
  v8f acc[4];
  float cmb[8];
  bool active = true;

  const bf16* xrow = X + (size_t)(rowBase + r) * Din + kl;

  issue_copy(0, 0);
  int buf = 0;
  for (int tix = 0; tix < ntiles; ++tix) {
    const int e     = MOE ? (tix / ksteps) : 0;
    const int kstep = tix - e * ksteps;
    const int k0    = kstep * 32;

    if (kstep == 0) {
      if (MOE) {
        float cv = combine[(size_t)(rowBase + r) * kE + e];
        active = (__ballot(cv != 0.0f) != 0);
        if (active) {
#pragma unroll
          for (int j = 0; j < 8; ++j)
            cmb[j] = combine[(size_t)(rowBase + j + hi) * kE + e];
        }
      }
#pragma unroll
      for (int s = 0; s < 4; ++s) acc[s] = zero8();
    }

    wait_async0();          // my async chunks of wtile[buf] have landed
    __syncthreads();        // -> everyone's have; prev reads of wtile[buf^1] are done
    if (tix + 1 < ntiles) issue_copy(tix + 1, buf ^ 1);

    if (!MOE || active) {
      v16bf aa = load_frag_bf16(xrow + k0);
      __builtin_prefetch(xrow + k0 + 64, 0, 1);       // global_prefetch_b8
#pragma unroll
      for (int s = 0; s < 4; ++s) {
        const bf16* lp = &wtile[buf][(colSub + s * 16 + r) * 32 + kl];
        v16bf bb = load_frag_bf16(lp);
        acc[s] = __builtin_amdgcn_wmma_f32_16x16x32_bf16(
            false, aa, false, bb, (short)0, acc[s], false, false);
      }
    }

    if (kstep == ksteps - 1) {
      if (MOE) {
        if (active) {
#pragma unroll
          for (int s = 0; s < 4; ++s)
#pragma unroll
            for (int j = 0; j < 8; ++j) out[s][j] += cmb[j] * acc[s][j];
        }
      } else {
#pragma unroll
        for (int s = 0; s < 4; ++s)
#pragma unroll
          for (int j = 0; j < 8; ++j) out[s][j] += acc[s][j];
      }
    }
    buf ^= 1;
  }

  // Epilogue. C layout: element j -> row = rowBase + j + hi, col = col0 + colSub + s*16 + r.
#pragma unroll
  for (int s = 0; s < 4; ++s) {
    const int col = col0 + colSub + s * 16 + r;
    const float bv = BIAS ? bias[col] : 0.0f;
#pragma unroll
    for (int j = 0; j < 8; ++j) {
      const int row = rowBase + j + hi;
      float v = out[s][j] + bv;
      if (GELU) {
        float u = v;
        v = 0.5f * u * (1.0f + tanhf(0.7978845608028654f * (u + 0.044715f * u * u * u)));
      }
      if (RES) v += resid[(size_t)row * Dout + col];
      if (OUT_BF16) Ybf[(size_t)row * Dout + col] = (bf16)v;
      else          Yf[(size_t)row * Dout + col]  = v;
    }
  }
}

// ---------------- Fallback WMMA GEMM (fp32 weights converted in-register) ----------------
template <bool MOE, bool BIAS, bool GELU, bool RES, bool OUT_BF16>
__global__ void __launch_bounds__(256)
wmma_gemm_f32w_kernel(const bf16* __restrict__ X, const float* __restrict__ W,
                      const float* __restrict__ combine, const float* __restrict__ bias,
                      const float* __restrict__ resid, float* __restrict__ Yf,
                      bf16* __restrict__ Ybf, int Din, int Dout, int nExp) {
  const int lane    = threadIdx.x & 31;
  const int wid     = threadIdx.x >> 5;
  const int rowBase = blockIdx.x * 64 + (wid & 3) * 16;
  const int colBase = blockIdx.y * 128 + (wid >> 2) * 64;
  const int r  = lane & 15;
  const int kl = (lane >> 4) << 3;
  const int hi = kl;

  v8f out[4];
#pragma unroll
  for (int s = 0; s < 4; ++s) out[s] = zero8();

  const int NE = MOE ? nExp : 1;
  for (int e = 0; e < NE; ++e) {
    float cmb[8];
    if (MOE) {
      float cv = combine[(size_t)(rowBase + r) * kE + e];
      if (__ballot(cv != 0.0f) == 0) continue;
#pragma unroll
      for (int j = 0; j < 8; ++j)
        cmb[j] = combine[(size_t)(rowBase + j + hi) * kE + e];
    }
    const float* We = W + (MOE ? (size_t)e * (size_t)Dout * (size_t)Din : (size_t)0);
    v8f acc[4];
#pragma unroll
    for (int s = 0; s < 4; ++s) acc[s] = zero8();

    const bf16* xrow = X + (size_t)(rowBase + r) * Din + kl;
    for (int k0 = 0; k0 < Din; k0 += 32) {
      v16bf aa = load_frag_bf16(xrow + k0);
#pragma unroll
      for (int s = 0; s < 4; ++s) {
        const float* wp = We + (size_t)(colBase + s * 16 + r) * Din + k0 + kl;
        v16bf bb = cvt_frag_f32(wp);
        acc[s] = __builtin_amdgcn_wmma_f32_16x16x32_bf16(
            false, aa, false, bb, (short)0, acc[s], false, false);
      }
    }
    if (MOE) {
#pragma unroll
      for (int s = 0; s < 4; ++s)
#pragma unroll
        for (int j = 0; j < 8; ++j) out[s][j] += cmb[j] * acc[s][j];
    } else {
#pragma unroll
      for (int s = 0; s < 4; ++s)
#pragma unroll
        for (int j = 0; j < 8; ++j) out[s][j] += acc[s][j];
    }
  }
#pragma unroll
  for (int s = 0; s < 4; ++s) {
    const int col = colBase + s * 16 + r;
    const float bv = BIAS ? bias[col] : 0.0f;
#pragma unroll
    for (int j = 0; j < 8; ++j) {
      const int row = rowBase + j + hi;
      float v = out[s][j] + bv;
      if (GELU) {
        float u = v;
        v = 0.5f * u * (1.0f + tanhf(0.7978845608028654f * (u + 0.044715f * u * u * u)));
      }
      if (RES) v += resid[(size_t)row * Dout + col];
      if (OUT_BF16) Ybf[(size_t)row * Dout + col] = (bf16)v;
      else          Yf[(size_t)row * Dout + col]  = v;
    }
  }
}

// ---------------- V transpose: [T, D] -> [B, H, HD, N] (bf16) for contiguous PV B-fragments ----
__global__ void vtrans_kernel(const bf16* __restrict__ V, bf16* __restrict__ Vt) {
  size_t o = (size_t)blockIdx.x * 256 + threadIdx.x;   // total = B*H*HD*N = 4M
  int n = (int)(o % kN); size_t q = o / kN;
  int d = (int)(q % kHD); q /= kHD;
  int h = (int)(q % kH);
  int b = (int)(q / kH);
  Vt[o] = V[((size_t)b * kN + n) * kD + h * kHD + d];
}

// ---------------- Flash attention: one wave per (b, h, 16-query tile), 32 keys per step ------
__global__ void __launch_bounds__(32)
attn_kernel(const bf16* __restrict__ Q, const bf16* __restrict__ Kb,
            const bf16* __restrict__ Vt, bf16* __restrict__ O) {
  __shared__ __align__(16) bf16 pl[16 * 32];   // P tile staging (C layout -> A layout)
  const int lane = threadIdx.x & 31;
  const int qt = blockIdx.x, h = blockIdx.y, b = blockIdx.z;
  const int r  = lane & 15;
  const int kl = (lane >> 4) << 3;
  const int hi = kl;

  const size_t qtok = (size_t)b * kN + qt * 16 + r;
  v16bf qa[2];
#pragma unroll
  for (int c = 0; c < 2; ++c)
    qa[c] = load_frag_bf16(Q + qtok * kD + h * kHD + c * 32 + kl);

  v8f o[4];
#pragma unroll
  for (int s = 0; s < 4; ++s) o[s] = zero8();
  float m[8], l[8];
#pragma unroll
  for (int j = 0; j < 8; ++j) { m[j] = -1e30f; l[j] = 0.f; }

  for (int kc = 0; kc < kN; kc += 32) {
    v8f s0 = zero8(), s1 = zero8();
#pragma unroll
    for (int c = 0; c < 2; ++c) {
      const bf16* kp0 = Kb + ((size_t)b * kN + kc + r) * kD + h * kHD + c * 32 + kl;
      v16bf b0 = load_frag_bf16(kp0);
      s0 = __builtin_amdgcn_wmma_f32_16x16x32_bf16(false, qa[c], false, b0, (short)0, s0, false, false);
      const bf16* kp1 = Kb + ((size_t)b * kN + kc + 16 + r) * kD + h * kHD + c * 32 + kl;
      v16bf b1 = load_frag_bf16(kp1);
      s1 = __builtin_amdgcn_wmma_f32_16x16x32_bf16(false, qa[c], false, b1, (short)0, s1, false, false);
    }
    float corr[8];
#pragma unroll
    for (int j = 0; j < 8; ++j) {
      float a0 = s0[j] * 0.125f, a1 = s1[j] * 0.125f;   // 1/sqrt(HD)
      float mx = fmaxf(a0, a1);
      for (int dlt = 1; dlt < 16; dlt <<= 1) mx = fmaxf(mx, __shfl_xor(mx, dlt, 32));
      float mn = fmaxf(m[j], mx);
      float e0 = __expf(a0 - mn), e1 = __expf(a1 - mn);
      float rs = e0 + e1;
      for (int dlt = 1; dlt < 16; dlt <<= 1) rs += __shfl_xor(rs, dlt, 32);
      corr[j] = __expf(m[j] - mn);
      l[j] = l[j] * corr[j] + rs;
      m[j] = mn;
      const int prow = j + hi;
      pl[prow * 32 + r]      = (bf16)e0;
      pl[prow * 32 + 16 + r] = (bf16)e1;
    }
    __syncthreads();
    asm volatile("s_wait_dscnt 0" ::: "memory");
    v16bf pa = load_frag_bf16(&pl[r * 32 + kl]);
#pragma unroll
    for (int s = 0; s < 4; ++s) {
#pragma unroll
      for (int j = 0; j < 8; ++j) o[s][j] *= corr[j];
      const bf16* vp = Vt + (((size_t)(b * kH + h)) * kHD + s * 16 + r) * kN + kc + kl;
      v16bf vb = load_frag_bf16(vp);
      o[s] = __builtin_amdgcn_wmma_f32_16x16x32_bf16(false, pa, false, vb, (short)0, o[s], false, false);
    }
    __syncthreads();
  }
#pragma unroll
  for (int s = 0; s < 4; ++s)
#pragma unroll
    for (int j = 0; j < 8; ++j) {
      int tok = b * kN + qt * 16 + j + hi;
      O[(size_t)tok * kD + h * kHD + s * 16 + r] = (bf16)(o[s][j] / l[j]);
    }
}

// ---------------- misc ----------------
__global__ void zero_kernel(float* p, int n) {
  int i = blockIdx.x * 256 + threadIdx.x;
  if (i < n) p[i] = 0.f;
}

__global__ void lb_kernel(const float* __restrict__ counts, const float* __restrict__ psum,
                          float* __restrict__ out) {
  if (threadIdx.x == 0 && blockIdx.x == 0) {
    float tot = 0.f;
#pragma unroll
    for (int e = 0; e < kE; ++e) tot += counts[e];
    float lb = 0.f;
#pragma unroll
    for (int e = 0; e < kE; ++e) lb += counts[e] / (tot + 1e-6f) * psum[e];
    out[(size_t)kT * kD] = lb * (float)kE;
  }
}

extern "C" void kernel_launch(void* const* d_in, const int* in_sizes, int n_in,
                              void* d_out, int out_size, void* d_ws, size_t ws_size,
                              hipStream_t stream) {
  (void)in_sizes; (void)n_in; (void)out_size;
  const float* x   = (const float*)d_in[0];
  const float* Wr  = (const float*)d_in[1];
  const float* Wq  = (const float*)d_in[2];
  const float* Wk  = (const float*)d_in[3];
  const float* Wv  = (const float*)d_in[4];
  const float* Wo  = (const float*)d_in[5];
  const float* W1  = (const float*)d_in[6];
  const float* b1  = (const float*)d_in[7];
  const float* W2  = (const float*)d_in[8];
  const float* b2  = (const float*)d_in[9];
  const float* g1  = (const float*)d_in[10];
  const float* be1 = (const float*)d_in[11];
  const float* g2  = (const float*)d_in[12];
  const float* be2 = (const float*)d_in[13];
  float* out = (float*)d_out;

  char* base = (char*)d_ws;
  size_t off = 0;
  auto take = [&](size_t bytes) {
    void* p = (void*)(base + off);
    off += (bytes + 255) & ~(size_t)255;
    return p;
  };
  // --- activation scratch ---
  bf16*  xnb    = (bf16*)take((size_t)kT * kD * sizeof(bf16));
  float* comb   = (float*)take((size_t)kT * kE * sizeof(float));
  float* cntps  = (float*)take((size_t)2 * kE * sizeof(float));
  float* counts = cntps;
  float* psum   = cntps + kE;
  bf16*  qb     = (bf16*)take((size_t)kT * kD * sizeof(bf16));
  bf16*  kb     = (bf16*)take((size_t)kT * kD * sizeof(bf16));
  bf16*  vb     = (bf16*)take((size_t)kT * kD * sizeof(bf16));
  bf16*  vt     = (bf16*)take((size_t)kT * kD * sizeof(bf16));
  bf16*  attnb  = (bf16*)take((size_t)kT * kD * sizeof(bf16));
  float* hbuf   = (float*)take((size_t)kT * kD * sizeof(float));
  bf16*  hnb    = (bf16*)take((size_t)kT * kD * sizeof(bf16));
  bf16*  ff1    = (bf16*)take((size_t)kT * kFF * sizeof(bf16));
  // --- bf16 weight cache (preferred path, gated on ws_size) ---
  const size_t nWqkvo = (size_t)kE * kD * kD;
  const size_t nW12   = (size_t)kFF * kD;
  bf16* wqb = (bf16*)take(nWqkvo * sizeof(bf16));
  bf16* wkb = (bf16*)take(nWqkvo * sizeof(bf16));
  bf16* wvb = (bf16*)take(nWqkvo * sizeof(bf16));
  bf16* wob = (bf16*)take(nWqkvo * sizeof(bf16));
  bf16* w1b = (bf16*)take(nW12 * sizeof(bf16));
  bf16* w2b = (bf16*)take(nW12 * sizeof(bf16));
  const bool use_wbf = (off <= ws_size);

  zero_kernel<<<1, 64, 0, stream>>>(cntps, 2 * kE);
  ln_bf16_kernel<<<kT, 256, 0, stream>>>(x, g1, be1, xnb);
  router_kernel<<<kT, 128, 0, stream>>>(xnb, Wr, comb, counts, psum);

  dim3 gD(kT / 64, kD / 128);     // (64, 8)
  dim3 gF(kT / 64, kFF / 128);    // (64, 32)
  dim3 ga(kN / 16, kH, kB);       // (64, 16, 4)

  if (use_wbf) {
    const int q4 = (int)(nWqkvo / 4), f4 = (int)(nW12 / 4);
    cvt_f32_bf16_kernel<<<(q4 + 255) / 256, 256, 0, stream>>>(Wq, wqb, q4);
    cvt_f32_bf16_kernel<<<(q4 + 255) / 256, 256, 0, stream>>>(Wk, wkb, q4);
    cvt_f32_bf16_kernel<<<(q4 + 255) / 256, 256, 0, stream>>>(Wv, wvb, q4);
    cvt_f32_bf16_kernel<<<(q4 + 255) / 256, 256, 0, stream>>>(Wo, wob, q4);
    cvt_f32_bf16_kernel<<<(f4 + 255) / 256, 256, 0, stream>>>(W1, w1b, f4);
    cvt_f32_bf16_kernel<<<(f4 + 255) / 256, 256, 0, stream>>>(W2, w2b, f4);

    wmma_gemm_bf16w_kernel<true, false, false, false, true><<<gD, 256, 0, stream>>>(
        xnb, wqb, comb, nullptr, nullptr, nullptr, qb, kD, kD, kE);
    wmma_gemm_bf16w_kernel<true, false, false, false, true><<<gD, 256, 0, stream>>>(
        xnb, wkb, comb, nullptr, nullptr, nullptr, kb, kD, kD, kE);
    wmma_gemm_bf16w_kernel<true, false, false, false, true><<<gD, 256, 0, stream>>>(
        xnb, wvb, comb, nullptr, nullptr, nullptr, vb, kD, kD, kE);
    vtrans_kernel<<<(kB * kH * kHD * kN) / 256, 256, 0, stream>>>(vb, vt);
    attn_kernel<<<ga, 32, 0, stream>>>(qb, kb, vt, attnb);
    wmma_gemm_bf16w_kernel<true, false, false, true, false><<<gD, 256, 0, stream>>>(
        attnb, wob, comb, nullptr, x, hbuf, nullptr, kD, kD, kE);
    ln_bf16_kernel<<<kT, 256, 0, stream>>>(hbuf, g2, be2, hnb);
    wmma_gemm_bf16w_kernel<false, true, true, false, true><<<gF, 256, 0, stream>>>(
        hnb, w1b, nullptr, b1, nullptr, nullptr, ff1, kD, kFF, 1);
    wmma_gemm_bf16w_kernel<false, true, false, true, false><<<gD, 256, 0, stream>>>(
        ff1, w2b, nullptr, b2, hbuf, out, nullptr, kFF, kD, 1);
  } else {
    wmma_gemm_f32w_kernel<true, false, false, false, true><<<gD, 256, 0, stream>>>(
        xnb, Wq, comb, nullptr, nullptr, nullptr, qb, kD, kD, kE);
    wmma_gemm_f32w_kernel<true, false, false, false, true><<<gD, 256, 0, stream>>>(
        xnb, Wk, comb, nullptr, nullptr, nullptr, kb, kD, kD, kE);
    wmma_gemm_f32w_kernel<true, false, false, false, true><<<gD, 256, 0, stream>>>(
        xnb, Wv, comb, nullptr, nullptr, nullptr, vb, kD, kD, kE);
    vtrans_kernel<<<(kB * kH * kHD * kN) / 256, 256, 0, stream>>>(vb, vt);
    attn_kernel<<<ga, 32, 0, stream>>>(qb, kb, vt, attnb);
    wmma_gemm_f32w_kernel<true, false, false, true, false><<<gD, 256, 0, stream>>>(
        attnb, Wo, comb, nullptr, x, hbuf, nullptr, kD, kD, kE);
    ln_bf16_kernel<<<kT, 256, 0, stream>>>(hbuf, g2, be2, hnb);
    wmma_gemm_f32w_kernel<false, true, true, false, true><<<gF, 256, 0, stream>>>(
        hnb, W1, nullptr, b1, nullptr, nullptr, ff1, kD, kFF, 1);
    wmma_gemm_f32w_kernel<false, true, false, true, false><<<gD, 256, 0, stream>>>(
        ff1, W2, nullptr, b2, hbuf, out, nullptr, kFF, kD, 1);
  }

  lb_kernel<<<1, 32, 0, stream>>>(counts, psum, out);
}
